// Time_AE_3126736191849
// MI455X (gfx1250) — compile-verified, matrix operand order checked by reading, not run on hardware
//
#include <hip/hip_runtime.h>
#include <hip/hip_bf16.h>

typedef __attribute__((ext_vector_type(16))) _Float16 v16h;
typedef __attribute__((ext_vector_type(8)))  float    v8f;

union FragH { v16h v; unsigned int u[8]; _Float16 e[16]; };
union FragF { v8f  v; float f[8]; };
union H2    { _Float16 h[2]; unsigned int u; };

// K-index map for 16-bit 16x32 A-frag / 32x16 B-frag (CDNA5 WMMA layout):
// lane group hl = lane>>4 ; vgpr j holds K pair (k0, k0+1), k0 always even
__device__ __forceinline__ int wmma_k0(int j, int hl) {
    return (j < 4) ? (2 * j + 8 * hl) : (16 + 2 * (j - 4) + 8 * hl);
}

// ---------------------------------------------------------------- f32 -> f16
__global__ void cvt_f32_f16_kernel(const float* __restrict__ in,
                                   _Float16* __restrict__ out, size_t n) {
    size_t i = (size_t)blockIdx.x * blockDim.x + threadIdx.x;
    if (i < n) out[i] = (_Float16)in[i];
}

// ---------------------------------------------------------------- embedding
__global__ void embed_kernel(const float* __restrict__ x,
                             const float* __restrict__ cw,
                             const float* __restrict__ cb,
                             float* __restrict__ dH,
                             _Float16* __restrict__ dH16,
                             int Bsz, int L) {
    size_t idx = (size_t)blockIdx.x * blockDim.x + threadIdx.x;
    size_t total = (size_t)Bsz * L * 512;
    if (idx >= total) return;
    int e = (int)(idx & 511);
    size_t ml = idx >> 9;
    int l = (int)(ml % L);
    int b = (int)(ml / L);
    float acc = cb[e];
    #pragma unroll
    for (int t = 0; t < 3; ++t) {
        int ls = l + t - 1;
        if (ls < 0) ls += L;
        if (ls >= L) ls -= L;
        const float* xp = x + ((size_t)b * L + ls) * 32;
        const float* wp = cw + (size_t)e * 96 + t;  // (e*32 + c)*3 + t
        #pragma unroll
        for (int c = 0; c < 32; ++c) acc += xp[c] * wp[c * 3];
    }
    int i2 = e & ~1;
    float dv = __expf(-(float)i2 * (9.210340371976184f / 512.0f));
    float ang = (float)l * dv;
    acc += (e & 1) ? cosf(ang) : sinf(ang);
    dH[idx] = acc;
    dH16[idx] = (_Float16)acc;
}

// ---------------------------------------------------------------- WMMA GEMM
// C[M,N] = A16[M,K] @ W16[N,K]^T + bias[N]
// One wave -> 16 rows x 64 cols (4 N-tiles). EP: 0=f32, 1=f16, 2=silu->f16
template <int EP>
__global__ __launch_bounds__(128) void gemm_wmma_kernel(
    const _Float16* __restrict__ A, const _Float16* __restrict__ W,
    const float* __restrict__ bias, void* __restrict__ Out,
    int M, int N, int K) {
    const int lane = threadIdx.x & 31;
    const int wave = threadIdx.x >> 5;
    const int hl = lane >> 4;
    const int r = lane & 15;
    const int mbase = (blockIdx.x * 4 + wave) * 16;
    const int nbase0 = blockIdx.y * 64;
    if (mbase >= M) return;

    FragF acc[4];
    #pragma unroll
    for (int t = 0; t < 4; ++t)
        #pragma unroll
        for (int i = 0; i < 8; ++i) acc[t].f[i] = 0.0f;

    const _Float16* arow = A + (size_t)(mbase + r) * K;
    for (int kb = 0; kb < K; kb += 32) {
        FragH a;
        #pragma unroll
        for (int j = 0; j < 8; ++j)
            a.u[j] = *(const unsigned int*)(arow + kb + wmma_k0(j, hl));
        #pragma unroll
        for (int t = 0; t < 4; ++t) {
            if (nbase0 + t * 16 < N) {
                const _Float16* wrow = W + (size_t)(nbase0 + t * 16 + r) * K;
                FragH bf;
                #pragma unroll
                for (int j = 0; j < 8; ++j)
                    bf.u[j] = *(const unsigned int*)(wrow + kb + wmma_k0(j, hl));
                acc[t].v = __builtin_amdgcn_wmma_f32_16x16x32_f16(
                    false, a.v, false, bf.v, (short)0, acc[t].v, false, false);
            }
        }
    }

    #pragma unroll
    for (int t = 0; t < 4; ++t) {
        int ntile = nbase0 + t * 16;
        if (ntile >= N) continue;
        int n = ntile + r;
        float bn = bias ? bias[n] : 0.0f;
        #pragma unroll
        for (int i = 0; i < 8; ++i) {
            int m = mbase + i + 8 * hl;
            float v = acc[t].f[i] + bn;
            if (EP == 2) v = v / (1.0f + __expf(-v));  // SiLU
            if (EP == 0)
                ((float*)Out)[(size_t)m * N + n] = v;
            else
                ((_Float16*)Out)[(size_t)m * N + n] = (_Float16)v;
        }
    }
}

// ---------------------------------------------------------------- attention
// qkv16: [B*L, 1536] f16 (Q | K | V, head-major). ctx16: [B*L, 512] f16.
// One wave per (b, head, 16-row q-tile). Flash softmax over 256-key chunks.
// V chunk staged in LDS pre-paired along K so every WMMA B-frag dword is one
// conflict-free ds_load_b32; P stored packed f16 so A-frags are ds_load_b32.
#define ACH 256
__global__ __launch_bounds__(32) void attn_kernel(
    const _Float16* __restrict__ qkv, _Float16* __restrict__ ctx, int L) {
    __shared__ __align__(16) float        S[16][ACH + 4];
    __shared__ __align__(16) _Float16     P[16][ACH + 10];
    __shared__ __align__(16) unsigned int Vp[ACH / 2][66]; // {V[2kk][d], V[2kk+1][d]}
    __shared__ float rowscale[16], rowsumS[16];

    const int lane = threadIdx.x;
    const int hl = lane >> 4;
    const int r = lane & 15;
    const int qbase = blockIdx.x * 16;
    const int hd = blockIdx.y;
    const size_t base = (size_t)blockIdx.z * L;

    // Q frag: A 16x64 -> two k-steps of 32 (cached in regs for whole kernel)
    FragH qa[2];
    {
        const _Float16* qrow = qkv + (base + qbase + r) * 1536 + hd * 64;
        #pragma unroll
        for (int s = 0; s < 2; ++s)
            #pragma unroll
            for (int j = 0; j < 8; ++j)
                qa[s].u[j] = *(const unsigned int*)(qrow + s * 32 + wmma_k0(j, hl));
    }

    FragF o[4];
    #pragma unroll
    for (int t = 0; t < 4; ++t)
        #pragma unroll
        for (int i = 0; i < 8; ++i) o[t].f[i] = 0.0f;
    float rmax = -3.0e38f, rsum = 0.0f;  // row stats for row (lane&15)

    for (int ch = 0; ch < L; ch += ACH) {
        // ---- stage V chunk into LDS, packed by key-pair -------------------
        #pragma unroll
        for (int it = 0; it < ACH / 64; ++it) {
            int kk = it * 32 + lane;  // key pair index
            const _Float16* v0 = qkv + (base + ch + 2 * kk) * 1536 + 1024 + hd * 64;
            #pragma unroll
            for (int d8 = 0; d8 < 8; ++d8) {
                union U4 { uint4 q; _Float16 h[8]; } a, b;
                a.q = *(const uint4*)(v0 + d8 * 8);
                b.q = *(const uint4*)(v0 + 1536 + d8 * 8);
                #pragma unroll
                for (int dd = 0; dd < 8; ++dd) {
                    H2 t;
                    t.h[0] = a.h[dd];
                    t.h[1] = b.h[dd];
                    Vp[kk][d8 * 8 + dd] = t.u;
                }
            }
        }
        // ---- S chunk = (Q @ K^T) * 1/sqrt(64) -----------------------------
        for (int kt = 0; kt < ACH; kt += 16) {
            FragF s;
            #pragma unroll
            for (int i = 0; i < 8; ++i) s.f[i] = 0.0f;
            const _Float16* krow = qkv + (base + ch + kt + r) * 1536 + 512 + hd * 64;
            #pragma unroll
            for (int st = 0; st < 2; ++st) {
                FragH kf;
                #pragma unroll
                for (int j = 0; j < 8; ++j)
                    kf.u[j] = *(const unsigned int*)(krow + st * 32 + wmma_k0(j, hl));
                s.v = __builtin_amdgcn_wmma_f32_16x16x32_f16(
                    false, qa[st].v, false, kf.v, (short)0, s.v, false, false);
            }
            #pragma unroll
            for (int i = 0; i < 8; ++i)
                S[i + 8 * hl][kt + r] = s.f[i] * 0.125f;
        }
        __syncthreads();
        // ---- online softmax: lanes r and r+16 split row r's columns -------
        {
            const int c0 = hl * (ACH / 2);
            float mc = rmax;
            for (int j = c0; j < c0 + ACH / 2; ++j) mc = fmaxf(mc, S[r][j]);
            mc = fmaxf(mc, __shfl_xor(mc, 16, 32));
            float alpha = __expf(rmax - mc);
            float ssum = 0.0f;
            for (int j = c0; j < c0 + ACH / 2; j += 2) {
                float p0 = __expf(S[r][j] - mc);
                float p1 = __expf(S[r][j + 1] - mc);
                ssum += p0 + p1;
                H2 t;
                t.h[0] = (_Float16)p0;
                t.h[1] = (_Float16)p1;
                *(unsigned int*)&P[r][j] = t.u;
            }
            ssum += __shfl_xor(ssum, 16, 32);
            rsum = rsum * alpha + ssum;
            rmax = mc;
            if (lane < 16) rowscale[lane] = alpha;
        }
        __syncthreads();
        // ---- rescale O, then O += P @ V -----------------------------------
        #pragma unroll
        for (int t = 0; t < 4; ++t)
            #pragma unroll
            for (int i = 0; i < 8; ++i) o[t].f[i] *= rowscale[i + 8 * hl];
        for (int kb = 0; kb < ACH; kb += 32) {
            FragH pa;
            #pragma unroll
            for (int j = 0; j < 8; ++j)
                pa.u[j] = *(const unsigned int*)&P[r][kb + wmma_k0(j, hl)];
            #pragma unroll
            for (int t = 0; t < 4; ++t) {
                FragH vf;
                #pragma unroll
                for (int j = 0; j < 8; ++j)
                    vf.u[j] = Vp[(kb + wmma_k0(j, hl)) >> 1][t * 16 + r];
                o[t].v = __builtin_amdgcn_wmma_f32_16x16x32_f16(
                    false, pa.v, false, vf.v, (short)0, o[t].v, false, false);
            }
        }
        __syncthreads();
    }
    if (lane < 16) rowsumS[lane] = rsum;
    __syncthreads();
    // ---- normalize + store -----------------------------------------------
    #pragma unroll
    for (int t = 0; t < 4; ++t)
        #pragma unroll
        for (int i = 0; i < 8; ++i) {
            float v = o[t].f[i] / rowsumS[i + 8 * hl];
            size_t row = base + qbase + i + 8 * hl;
            ctx[row * 512 + hd * 64 + t * 16 + r] = (_Float16)v;
        }
}

// ---------------------------------------------------------------- residual+LN
__global__ __launch_bounds__(256) void resln_kernel(
    float* __restrict__ dH, const float* __restrict__ dT,
    const float* __restrict__ w, const float* __restrict__ bb,
    _Float16* __restrict__ dH16, int M) {
    int row = blockIdx.x * 8 + (threadIdx.x >> 5);
    int lane = threadIdx.x & 31;
    if (row >= M) return;
    const size_t off = (size_t)row * 512;
    float x[16];
    float sum = 0.0f;
    #pragma unroll
    for (int i = 0; i < 16; ++i) {
        x[i] = dH[off + lane + 32 * i] + dT[off + lane + 32 * i];
        sum += x[i];
    }
    #pragma unroll
    for (int d = 16; d >= 1; d >>= 1) sum += __shfl_xor(sum, d, 32);
    float mu = sum * (1.0f / 512.0f);
    float var = 0.0f;
    #pragma unroll
    for (int i = 0; i < 16; ++i) { x[i] -= mu; var += x[i] * x[i]; }
    #pragma unroll
    for (int d = 16; d >= 1; d >>= 1) var += __shfl_xor(var, d, 32);
    float inv = rsqrtf(var * (1.0f / 512.0f) + 1e-5f);
    #pragma unroll
    for (int i = 0; i < 16; ++i) {
        int c = lane + 32 * i;
        float y = x[i] * inv * w[c] + bb[c];
        dH[off + c] = y;
        dH16[off + c] = (_Float16)y;
    }
}

// ---------------------------------------------------------------- launch
extern "C" void kernel_launch(void* const* d_in, const int* in_sizes, int n_in,
                              void* d_out, int out_size, void* d_ws, size_t ws_size,
                              hipStream_t stream) {
    const int Bsz = 4, L = 2048, E = 512, NL = 4;
    const int M = Bsz * L;  // 8192

    const float* x      = (const float*)d_in[0];
    const float* conv_w = (const float*)d_in[1];
    const float* conv_b = (const float*)d_in[2];
    const float* qkv_w  = (const float*)d_in[3];
    const float* qkv_b  = (const float*)d_in[4];
    const float* out_w  = (const float*)d_in[5];
    const float* out_b  = (const float*)d_in[6];
    const float* fc1_w  = (const float*)d_in[7];
    const float* fc1_b  = (const float*)d_in[8];
    const float* fc2_w  = (const float*)d_in[9];
    const float* fc2_b  = (const float*)d_in[10];
    const float* ln1_w  = (const float*)d_in[11];
    const float* ln1_b  = (const float*)d_in[12];
    const float* ln2_w  = (const float*)d_in[13];
    const float* ln2_b  = (const float*)d_in[14];
    const float* outp_w = (const float*)d_in[15];
    const float* outp_b = (const float*)d_in[16];

    char* ws = (char*)d_ws;
    size_t off = 0;
    auto alloc = [&](size_t bytes) { char* p = ws + off; off += (bytes + 255) & ~(size_t)255; return p; };
    float*    dH     = (float*)alloc((size_t)M * E * 4);
    float*    dT     = (float*)alloc((size_t)M * E * 4);
    _Float16* dH16   = (_Float16*)alloc((size_t)M * E * 2);
    _Float16* dA16   = (_Float16*)alloc((size_t)M * E * 2);
    _Float16* dCTX16 = (_Float16*)alloc((size_t)M * E * 2);
    _Float16* dQKV16 = (_Float16*)alloc((size_t)M * 1536 * 2);
    _Float16* wQKV   = (_Float16*)alloc((size_t)NL * 1536 * 512 * 2);
    _Float16* wOUT   = (_Float16*)alloc((size_t)NL * 512 * 512 * 2);
    _Float16* wFC1   = (_Float16*)alloc((size_t)NL * 512 * 512 * 2);
    _Float16* wFC2   = (_Float16*)alloc((size_t)NL * 512 * 512 * 2);
    _Float16* wOUTP  = (_Float16*)alloc((size_t)32 * 512 * 2);

    auto cvt = [&](const float* src, _Float16* dst, size_t n) {
        cvt_f32_f16_kernel<<<(unsigned)((n + 255) / 256), 256, 0, stream>>>(src, dst, n);
    };
    cvt(qkv_w, wQKV, (size_t)NL * 1536 * 512);
    cvt(out_w, wOUT, (size_t)NL * 512 * 512);
    cvt(fc1_w, wFC1, (size_t)NL * 512 * 512);
    cvt(fc2_w, wFC2, (size_t)NL * 512 * 512);
    cvt(outp_w, wOUTP, (size_t)32 * 512);

    {
        size_t n = (size_t)M * E;
        embed_kernel<<<(unsigned)((n + 255) / 256), 256, 0, stream>>>(
            x, conv_w, conv_b, dH, dH16, Bsz, L);
    }

    for (int l = 0; l < NL; ++l) {
        gemm_wmma_kernel<1><<<dim3(M / 64, 1536 / 64), 128, 0, stream>>>(
            dH16, wQKV + (size_t)l * 1536 * 512, qkv_b + (size_t)l * 1536,
            dQKV16, M, 1536, 512);
        attn_kernel<<<dim3(L / 16, 8, Bsz), 32, 0, stream>>>(dQKV16, dCTX16, L);
        gemm_wmma_kernel<0><<<dim3(M / 64, 512 / 64), 128, 0, stream>>>(
            dCTX16, wOUT + (size_t)l * 512 * 512, out_b + (size_t)l * 512,
            dT, M, 512, 512);
        resln_kernel<<<M / 8, 256, 0, stream>>>(
            dH, dT, ln1_w + (size_t)l * 512, ln1_b + (size_t)l * 512, dH16, M);
        gemm_wmma_kernel<2><<<dim3(M / 64, 512 / 64), 128, 0, stream>>>(
            dH16, wFC1 + (size_t)l * 512 * 512, fc1_b + (size_t)l * 512,
            dA16, M, 512, 512);
        gemm_wmma_kernel<0><<<dim3(M / 64, 512 / 64), 128, 0, stream>>>(
            dA16, wFC2 + (size_t)l * 512 * 512, fc2_b + (size_t)l * 512,
            dT, M, 512, 512);
        resln_kernel<<<M / 8, 256, 0, stream>>>(
            dH, dT, ln2_w + (size_t)l * 512, ln2_b + (size_t)l * 512, dH16, M);
    }

    gemm_wmma_kernel<0><<<dim3(M / 64, 1), 128, 0, stream>>>(
        dH16, wOUTP, outp_b, d_out, M, 32, 512);
    (void)in_sizes; (void)n_in; (void)out_size; (void)ws_size;
}
